// Block_27384711479862
// MI455X (gfx1250) — matrix operand order repaired and checked by reading, not compiled, not run on hardware
//
#include <hip/hip_runtime.h>
#include <hip/hip_bf16.h>
#include <math.h>

#define BATCH 8
#define CDIM  384
#define NPOS  2304
#define HIDD  1536
#define HW    48

typedef __attribute__((ext_vector_type(16))) _Float16 v16h;
typedef __attribute__((ext_vector_type(8)))  float    v8f;

// ---------------------------------------------------------------------------
// WMMA helpers (CDNA5 gfx1250, wave32, 16x16x32 f16 -> f32)
// Convention: D[M][N] = A[M][K] * Bt[N][K]^T, with both operands stored
// K-contiguous (A row-major, B transposed row-major). Fragment loads follow
// the ISA 16-bit A (16x32) and B (32x16) VGPR layouts (CDNA5 ISA).
// ---------------------------------------------------------------------------
__device__ __forceinline__ v8f wmma_f16(v16h a, v16h b, v8f c) {
  return __builtin_amdgcn_wmma_f32_16x16x32_f16(false, a, false, b, (short)0, c,
                                                false, false);
}

// A fragment 16x32: lane half h holds K = {8h..8h+7, 16+8h..16+8h+7}
__device__ __forceinline__ v16h load_afrag(const _Float16* p, int ld) {
  const int lane = threadIdx.x & 31;
  const int r = lane & 15, h = lane >> 4;
  const _Float16* q = p + (size_t)r * ld + h * 8;
  union { v16h v; float4 f[2]; } u;
  u.f[0] = *(const float4*)(q);
  u.f[1] = *(const float4*)(q + 16);
  return u.v;
}

// B fragment 32x16 (from Bt[N][K]): lane holds col n = lane&15,
// K = {8h..8h+7, 16+8h..16+8h+7}, h = lane>>4.
__device__ __forceinline__ v16h load_bfrag(const _Float16* p, int ld) {
  const int lane = threadIdx.x & 31;
  const int n = lane & 15, h = lane >> 4;
  const _Float16* q = p + (size_t)n * ld + h * 8;
  union { v16h v; float4 f[2]; } u;
  u.f[0] = *(const float4*)(q);
  u.f[1] = *(const float4*)(q + 16);
  return u.v;
}

// ---------------------------------------------------------------------------
// f32 -> f16 weight conversion
// ---------------------------------------------------------------------------
__global__ void cvt_f32_f16(const float* __restrict__ s,
                            _Float16* __restrict__ d, int n) {
  int i = blockIdx.x * blockDim.x + threadIdx.x;
  if (i < n) d[i] = (_Float16)s[i];
}

// ---------------------------------------------------------------------------
// BatchNorm (eval) + transpose [C][N] f32 -> [N][C] f16 via LDS tile
// ---------------------------------------------------------------------------
__global__ __launch_bounds__(256) void bnpack_kernel(
    const float* __restrict__ src, const float* __restrict__ gam,
    const float* __restrict__ bet, const float* __restrict__ mu,
    const float* __restrict__ var, _Float16* __restrict__ dst) {
  __shared__ float tile[32][33];
  const int n0 = blockIdx.x * 32, c0 = blockIdx.y * 32;
  const int tx = threadIdx.x & 31, ty = threadIdx.x >> 5;  // 32 x 8
  #pragma unroll
  for (int i = 0; i < 32; i += 8) {
    const int c = c0 + ty + i;
    const float inv = rsqrtf(var[c] + 1e-5f);
    const float x = src[(size_t)c * NPOS + n0 + tx];
    tile[ty + i][tx] = (x - mu[c]) * (gam[c] * inv) + bet[c];
  }
  __syncthreads();
  #pragma unroll
  for (int i = 0; i < 32; i += 8) {
    const int n = n0 + ty + i;
    dst[(size_t)n * CDIM + c0 + tx] = (_Float16)tile[tx][ty + i];
  }
}

// ---------------------------------------------------------------------------
// Generic WMMA GEMM: 256 threads = 8 waves, block tile 128(M) x 64(N),
// wave tile 32x32 (2x2 WMMA tiles), K-loop step 32.
// MODE 0: outH[m][n] = (acc + bias[n]) * scale            (f16)
// MODE 1: outH[m][n] = acc + bias[m]                      (f16, v: [C][N])
// MODE 2: proj+res1:  outF[n][m] = resX[n][m] + ls[n]*(acc + bias[n] + addH[m][n])
// MODE 3: fc2+res2:   outF[n][m] += ls[n]*(acc + bias[n])
// ---------------------------------------------------------------------------
template <int MODE>
__global__ __launch_bounds__(256) void gemm_kernel(
    const _Float16* __restrict__ A, const _Float16* __restrict__ Bt,
    int M, int Nc, int K, const float* __restrict__ bias, float scale,
    _Float16* __restrict__ outH, float* __restrict__ outF,
    const float* __restrict__ resX, const _Float16* __restrict__ addH,
    const float* __restrict__ ls) {
  const int wave = threadIdx.x >> 5;
  const int lane = threadIdx.x & 31;
  const int col = lane & 15, half = lane >> 4;
  const int m0 = blockIdx.x * 128 + (wave >> 1) * 32;
  const int n0 = blockIdx.y * 64 + (wave & 1) * 32;

  v8f acc[2][2] = {};
  for (int k0 = 0; k0 < K; k0 += 32) {
    v16h a0 = load_afrag(A + (size_t)m0 * K + k0, K);
    v16h a1 = load_afrag(A + (size_t)(m0 + 16) * K + k0, K);
    v16h b0 = load_bfrag(Bt + (size_t)n0 * K + k0, K);
    v16h b1 = load_bfrag(Bt + (size_t)(n0 + 16) * K + k0, K);
    acc[0][0] = wmma_f16(a0, b0, acc[0][0]);
    acc[0][1] = wmma_f16(a0, b1, acc[0][1]);
    acc[1][0] = wmma_f16(a1, b0, acc[1][0]);
    acc[1][1] = wmma_f16(a1, b1, acc[1][1]);
  }
  #pragma unroll
  for (int i = 0; i < 2; ++i)
    #pragma unroll
    for (int j = 0; j < 2; ++j) {
      const int mb = m0 + 16 * i + 8 * half;
      const int nc = n0 + 16 * j + col;
      #pragma unroll
      for (int r = 0; r < 8; ++r) {
        const int mr = mb + r;
        const float va = acc[i][j][r];
        if constexpr (MODE == 0) {
          outH[(size_t)mr * Nc + nc] = (_Float16)((va + bias[nc]) * scale);
        } else if constexpr (MODE == 1) {
          outH[(size_t)mr * Nc + nc] = (_Float16)(va + bias[mr]);
        } else if constexpr (MODE == 2) {
          const float a2 = va + bias[nc] + (float)addH[(size_t)mr * Nc + nc];
          outF[(size_t)nc * M + mr] = resX[(size_t)nc * M + mr] + ls[nc] * a2;
        } else {
          outF[(size_t)nc * M + mr] += ls[nc] * (va + bias[nc]);
        }
      }
    }
}

// ---------------------------------------------------------------------------
// Flash attention, one 16-query tile per 128-thread (4-wave) block.
// Q,K: [N][C] f16 (q pre-scaled by 1/sqrt(C)).  V: [C][N] f16.  O: [N][C] f16.
// Per 64-key chunk: each wave computes one 16x16 S tile (12 WMMAs over C=384),
// online-softmax stats via 16-lane shuffles + LDS, P staged in LDS f16 and
// consumed as B-fragments of O^T = V * P^T (each wave owns 96 channels).
// ---------------------------------------------------------------------------
__global__ __launch_bounds__(128) void attn_kernel(
    const _Float16* __restrict__ Q, const _Float16* __restrict__ Kmat,
    const _Float16* __restrict__ Vmat, _Float16* __restrict__ O) {
  __shared__ __align__(16) _Float16 Pbuf[16][64];
  __shared__ float smax[4][16];
  __shared__ float ssum[4][16];
  __shared__ float s_m[16], s_l[16], s_scale[16];

  const int wave = threadIdx.x >> 5;
  const int lane = threadIdx.x & 31;
  const int col = lane & 15, half = lane >> 4;
  const int q0 = blockIdx.x * 16;
  const int cbase = wave * 96;

  if (threadIdx.x < 16) { s_m[threadIdx.x] = -1e30f; s_l[threadIdx.x] = 0.0f; }
  __syncthreads();

  v8f oacc[6] = {};

  for (int key0 = 0; key0 < NPOS; key0 += 64) {
    // ---- S tile: queries q0..q0+15 x keys kb..kb+15 -------------------------
    const int kb = key0 + 16 * wave;
    v8f s = {};
    #pragma unroll
    for (int c = 0; c < CDIM; c += 32) {
      v16h af = load_afrag(Q + (size_t)q0 * CDIM + c, CDIM);
      v16h bf = load_bfrag(Kmat + (size_t)kb * CDIM + c, CDIM);
      s = wmma_f16(af, bf, s);
    }
    // ---- per-row max over this wave's 16 key-columns ------------------------
    #pragma unroll
    for (int r = 0; r < 8; ++r) {
      float mv = s[r];
      mv = fmaxf(mv, __shfl_xor(mv, 1, 16));
      mv = fmaxf(mv, __shfl_xor(mv, 2, 16));
      mv = fmaxf(mv, __shfl_xor(mv, 4, 16));
      mv = fmaxf(mv, __shfl_xor(mv, 8, 16));
      if (col == r) smax[wave][8 * half + r] = mv;
    }
    __syncthreads();
    // ---- combined new running max -------------------------------------------
    float mnew[8];
    #pragma unroll
    for (int r = 0; r < 8; ++r) {
      const int row = 8 * half + r;
      const float cm = fmaxf(fmaxf(smax[0][row], smax[1][row]),
                             fmaxf(smax[2][row], smax[3][row]));
      mnew[r] = fmaxf(s_m[row], cm);
    }
    // ---- P = exp(s - m_new); partial row sums; stage P in LDS ---------------
    #pragma unroll
    for (int r = 0; r < 8; ++r) {
      const float p = __expf(s[r] - mnew[r]);
      float sv = p;
      sv += __shfl_xor(sv, 1, 16);
      sv += __shfl_xor(sv, 2, 16);
      sv += __shfl_xor(sv, 4, 16);
      sv += __shfl_xor(sv, 8, 16);
      if (col == r) ssum[wave][8 * half + r] = sv;
      Pbuf[8 * half + r][16 * wave + col] = (_Float16)p;
    }
    __syncthreads();
    // ---- stats update (rows 0..15 handled by first 16 threads) --------------
    if (threadIdx.x < 16) {
      const int row = threadIdx.x;
      const float cm = fmaxf(fmaxf(smax[0][row], smax[1][row]),
                             fmaxf(smax[2][row], smax[3][row]));
      const float mn = fmaxf(s_m[row], cm);
      const float sc = __expf(s_m[row] - mn);
      s_scale[row] = sc;
      s_l[row] = s_l[row] * sc +
                 (ssum[0][row] + ssum[1][row] + ssum[2][row] + ssum[3][row]);
      s_m[row] = mn;
    }
    __syncthreads();
    // ---- O^T += V * P^T over this 64-key chunk ------------------------------
    const float sc = s_scale[col];          // per-query rescale (q = col)
    v16h pf0 = load_bfrag(&Pbuf[0][0], 64);  // keys key0 .. +31
    v16h pf1 = load_bfrag(&Pbuf[0][32], 64); // keys key0+32 .. +63
    #pragma unroll
    for (int t = 0; t < 6; ++t) {
      #pragma unroll
      for (int r = 0; r < 8; ++r) oacc[t][r] *= sc;
      v16h va0 = load_afrag(Vmat + (size_t)(cbase + 16 * t) * NPOS + key0, NPOS);
      v16h va1 = load_afrag(Vmat + (size_t)(cbase + 16 * t) * NPOS + key0 + 32, NPOS);
      oacc[t] = wmma_f16(va0, pf0, oacc[t]);
      oacc[t] = wmma_f16(va1, pf1, oacc[t]);
    }
    __syncthreads();  // protect Pbuf/smax reuse next chunk
  }
  // ---- normalize and write O[q][c] ------------------------------------------
  const float linv = 1.0f / s_l[col];
  #pragma unroll
  for (int t = 0; t < 6; ++t)
    #pragma unroll
    for (int r = 0; r < 8; ++r) {
      const int c = cbase + 16 * t + 8 * half + r;
      O[(size_t)(q0 + col) * CDIM + c] = (_Float16)(oacc[t][r] * linv);
    }
}

// ---------------------------------------------------------------------------
// Depthwise 3x3 (SAME, zero pad) + exact GELU.  h,g: [N][Hd] f16.
// ---------------------------------------------------------------------------
__global__ __launch_bounds__(256) void dwgelu_kernel(
    const _Float16* __restrict__ h, const float* __restrict__ w9,
    const float* __restrict__ bias, _Float16* __restrict__ g) {
  const int j = blockIdx.x * 256 + threadIdx.x;
  const int n = blockIdx.y;
  const int y = n / HW, x = n % HW;
  float acc = bias[j];
  #pragma unroll
  for (int dy = -1; dy <= 1; ++dy) {
    const int yy = y + dy;
    if (yy < 0 || yy >= HW) continue;
    #pragma unroll
    for (int dx = -1; dx <= 1; ++dx) {
      const int xx = x + dx;
      if (xx < 0 || xx >= HW) continue;
      acc += w9[j * 9 + (dy + 1) * 3 + (dx + 1)] *
             (float)h[(size_t)(yy * HW + xx) * HIDD + j];
    }
  }
  const float ge = 0.5f * acc * (1.0f + erff(acc * 0.70710678118654752f));
  g[(size_t)n * HIDD + j] = (_Float16)ge;
}

// ---------------------------------------------------------------------------
// Host orchestration
// ---------------------------------------------------------------------------
extern "C" void kernel_launch(void* const* d_in, const int* in_sizes, int n_in,
                              void* d_out, int out_size, void* d_ws,
                              size_t ws_size, hipStream_t stream) {
  (void)in_sizes; (void)n_in; (void)out_size; (void)ws_size;
  const float* x    = (const float*)d_in[0];
  const float* bn1g = (const float*)d_in[1];
  const float* bn1b = (const float*)d_in[2];
  const float* bn1m = (const float*)d_in[3];
  const float* bn1v = (const float*)d_in[4];
  const float* bn2g = (const float*)d_in[5];
  const float* bn2b = (const float*)d_in[6];
  const float* bn2m = (const float*)d_in[7];
  const float* bn2v = (const float*)d_in[8];
  const float* qw   = (const float*)d_in[9];
  const float* qbs  = (const float*)d_in[10];
  const float* kw   = (const float*)d_in[11];
  const float* kbs  = (const float*)d_in[12];
  const float* vw   = (const float*)d_in[13];
  const float* vbs  = (const float*)d_in[14];
  const float* pow_ = (const float*)d_in[15];
  const float* pob  = (const float*)d_in[16];
  const float* ls1  = (const float*)d_in[17];
  const float* ls2  = (const float*)d_in[18];
  const float* f1w  = (const float*)d_in[19];
  const float* f1b  = (const float*)d_in[20];
  const float* dww  = (const float*)d_in[21];
  const float* dwb  = (const float*)d_in[22];
  const float* f2w  = (const float*)d_in[23];
  const float* f2b  = (const float*)d_in[24];
  float* out = (float*)d_out;

  char* wsb = (char*)d_ws;
  size_t off = 0;
  auto halloc = [&](size_t elems) -> _Float16* {
    _Float16* p = (_Float16*)(wsb + off);
    off += (elems * sizeof(_Float16) + 255) & ~(size_t)255;
    return p;
  };
  const size_t CC = (size_t)CDIM * CDIM;
  const size_t HC = (size_t)HIDD * CDIM;
  const size_t NC = (size_t)NPOS * CDIM;
  const size_t NH = (size_t)NPOS * HIDD;

  _Float16* wqh = halloc(CC);
  _Float16* wkh = halloc(CC);
  _Float16* wvh = halloc(CC);
  _Float16* wph = halloc(CC);
  _Float16* w1h = halloc(HC);
  _Float16* w2h = halloc(HC);
  _Float16* xb1 = halloc(NC);   // bn1(x)  [N][C]
  _Float16* qb  = halloc(NC);   // q       [N][C] (pre-scaled)
  _Float16* kb  = halloc(NC);   // k       [N][C]
  _Float16* vb  = halloc(NC);   // v       [C][N]
  _Float16* ob  = halloc(NC);   // attn out [N][C]
  _Float16* xb2 = halloc(NC);   // bn2(x1) [N][C]
  _Float16* hb  = halloc(NH);   // fc1 out [N][Hd]
  _Float16* gb  = halloc(NH);   // gelu(dw) [N][Hd]

  // Weight conversion (once)
  cvt_f32_f16<<<(int)((CC + 255) / 256), 256, 0, stream>>>(qw, wqh, (int)CC);
  cvt_f32_f16<<<(int)((CC + 255) / 256), 256, 0, stream>>>(kw, wkh, (int)CC);
  cvt_f32_f16<<<(int)((CC + 255) / 256), 256, 0, stream>>>(vw, wvh, (int)CC);
  cvt_f32_f16<<<(int)((CC + 255) / 256), 256, 0, stream>>>(pow_, wph, (int)CC);
  cvt_f32_f16<<<(int)((HC + 255) / 256), 256, 0, stream>>>(f1w, w1h, (int)HC);
  cvt_f32_f16<<<(int)((HC + 255) / 256), 256, 0, stream>>>(f2w, w2h, (int)HC);

  const float qscale = 0.051031036307982884f;  // 1/sqrt(384)
  const dim3 gNC(NPOS / 128, CDIM / 64);       // M=N, Nc=C
  const dim3 gCN(CDIM / 128, NPOS / 64);       // M=C, Nc=N
  const dim3 gNH(NPOS / 128, HIDD / 64);       // M=N, Nc=Hd
  const dim3 gBN(NPOS / 32, CDIM / 32);

  for (int b = 0; b < BATCH; ++b) {
    const float* xbp = x + (size_t)b * CDIM * NPOS;   // [C][N]
    float* obp = out + (size_t)b * CDIM * NPOS;       // [C][N]

    bnpack_kernel<<<gBN, 256, 0, stream>>>(xbp, bn1g, bn1b, bn1m, bn1v, xb1);

    gemm_kernel<0><<<gNC, 256, 0, stream>>>(xb1, wqh, NPOS, CDIM, CDIM, qbs,
                                            qscale, qb, nullptr, nullptr,
                                            nullptr, nullptr);
    gemm_kernel<0><<<gNC, 256, 0, stream>>>(xb1, wkh, NPOS, CDIM, CDIM, kbs,
                                            1.0f, kb, nullptr, nullptr,
                                            nullptr, nullptr);
    gemm_kernel<1><<<gCN, 256, 0, stream>>>(wvh, xb1, CDIM, NPOS, CDIM, vbs,
                                            1.0f, vb, nullptr, nullptr,
                                            nullptr, nullptr);

    attn_kernel<<<NPOS / 16, 128, 0, stream>>>(qb, kb, vb, ob);

    // proj + (a + xb1) residual + layer-scale 1, writes NCHW f32
    gemm_kernel<2><<<gNC, 256, 0, stream>>>(ob, wph, NPOS, CDIM, CDIM, pob,
                                            1.0f, nullptr, obp, xbp, xb1, ls1);

    bnpack_kernel<<<gBN, 256, 0, stream>>>(obp, bn2g, bn2b, bn2m, bn2v, xb2);

    gemm_kernel<0><<<gNH, 256, 0, stream>>>(xb2, w1h, NPOS, HIDD, CDIM, f1b,
                                            1.0f, hb, nullptr, nullptr,
                                            nullptr, nullptr);

    dwgelu_kernel<<<dim3(HIDD / 256, NPOS), 256, 0, stream>>>(hb, dww, dwb, gb);

    // fc2 + layer-scale 2, accumulates into NCHW f32 output
    gemm_kernel<3><<<gNC, 256, 0, stream>>>(gb, w2h, NPOS, CDIM, HIDD, f2b,
                                            1.0f, nullptr, obp, nullptr,
                                            nullptr, ls2);
  }
}
